// MLPperagent_9139690406009
// MI455X (gfx1250) — matrix operand order repaired and checked by reading, not compiled, not run on hardware
//
#include <hip/hip_runtime.h>

// ---------------------------------------------------------------------------
// Per-agent MLP ensemble for MI455X (gfx1250, wave32, WMMA).
//   x = state_seq[:, -1, :]            (1024 x 256)  f32
//   h1 = relu(x  @ W1[a] + b1[a])      (1024 x 512)  per agent a in [0,64)
//   h2 = relu(h1 @ W2[a] + b2[a])      (1024 x 512)
//   h3 = relu(h2 @ W3[a] + b3[a])      (1024 x 512)
//   out[:,a,:] = h3 @ W4[a] + b4[a]    (1024 x 16)
//
// v2b: weights pre-converted to bf16 + pre-transposed into a chunk-major,
// fragment-native layout in d_ws by a prepass kernel. The GEMM kernel then
// stages each 40KB weight chunk with double-buffered
// global_load_async_to_lds_b128 (ASYNCcnt-tracked), overlapped with
// v_wmma_f32_16x16x32_bf16 compute. Activations stay in LDS for all layers.
// ---------------------------------------------------------------------------

typedef __bf16 bf16_t;
typedef __bf16 v16bf __attribute__((ext_vector_type(16)));
typedef __bf16 v8bf  __attribute__((ext_vector_type(8)));
typedef float  v8f   __attribute__((ext_vector_type(8)));
typedef float  v4f   __attribute__((ext_vector_type(4)));
typedef int    v4i   __attribute__((vector_size(16)));   // matches builtin param
typedef __attribute__((address_space(1))) v4i* gptr_b128;  // global (AS1)
typedef __attribute__((address_space(3))) v4i* lptr_b128;  // LDS (AS3)

#define NB        1024   // batch
#define SEQ       32     // sequence length (only last element used)
#define DIN       256    // state dim
#define NAGENT    64
#define HID       512
#define KOUT      16
#define BM        64     // batch rows per block
#define NTHREADS  256    // 8 waves (wave32)
#define KT        32     // K per chunk == one WMMA K step
#define KTP       40     // padded per-column stride (80B: 16B-aligned, conflict-free)
#define CHUNK_OPS 10     // b128 async ops per thread per 512-col chunk (512*40*2/256/16)

#if defined(__AMDGCN__) && __has_builtin(__builtin_amdgcn_global_load_async_to_lds_b128) && \
    __has_builtin(__builtin_amdgcn_s_wait_asynccnt)
#define USE_ASYNC_LDS 1
#else
#define USE_ASYNC_LDS 0
#endif

// --------------------------- fragment loaders ------------------------------
// A-fragment (16x32 bf16, MxK): lane<16 holds K[k0..k0+7] + K[k0+16..k0+23],
// lane>=16 holds K[k0+8..k0+15] + K[k0+24..k0+31]; p already includes khalf*8.
static __device__ __forceinline__ v16bf load_frag_a(const bf16_t* p) {
  v8bf lo = *(const v8bf*)(p);
  v8bf hi = *(const v8bf*)(p + 16);
  return __builtin_shufflevector(lo, hi, 0,1,2,3,4,5,6,7,8,9,10,11,12,13,14,15);
}
// B-fragment (32x16 bf16, KxN): lane<16 holds K[0..7] + K[16..23] of its column,
// lane>=16 holds K[8..15] + K[24..31]; p already includes khalf*8 within the
// transposed (column-major) weight column.
static __device__ __forceinline__ v16bf load_frag_b(const bf16_t* p) {
  v8bf lo = *(const v8bf*)(p);
  v8bf hi = *(const v8bf*)(p + 16);
  return __builtin_shufflevector(lo, hi, 0,1,2,3,4,5,6,7,8,9,10,11,12,13,14,15);
}

// --------------------- async (or sync) chunk staging -----------------------
// Chunk = HID columns x KTP bf16 = 40960 contiguous bytes in both global+LDS.
static __device__ __forceinline__ void stage_chunk(const bf16_t* __restrict__ src,
                                                   bf16_t* __restrict__ lds, int tid) {
  const char* s = (const char*)src;
  char*       l = (char*)lds;
#if USE_ASYNC_LDS
#pragma unroll
  for (int i = 0; i < CHUNK_OPS; ++i) {
    const int off = tid * 16 + i * (NTHREADS * 16);
    __builtin_amdgcn_global_load_async_to_lds_b128(
        (gptr_b128)(s + off), (lptr_b128)(l + off), 0, 0);
  }
#else
#pragma unroll
  for (int i = 0; i < CHUNK_OPS; ++i) {
    const int off = tid * 16 + i * (NTHREADS * 16);
    *(v8bf*)(l + off) = *(const v8bf*)(s + off);
  }
#endif
}

// ------------------------ prepass: convert+transpose -----------------------
// W (f32, row-major [KDIM][NDIM] per agent) -> dst (bf16, chunk-major:
// for chunk c (k = 32c..32c+31):  dst[((c*NDIM)+n)*KTP + (k-32c)] )
template<int KDIM, int NDIM>
__global__ __launch_bounds__(NTHREADS)
void convert_weights_kernel(const float* __restrict__ W, bf16_t* __restrict__ dst) {
  const int c     = blockIdx.x;   // k-chunk
  const int agent = blockIdx.y;
  const float* Wa = W + (long)agent * KDIM * NDIM + (long)c * KT * NDIM;
  bf16_t*      da = dst + (long)agent * (KDIM / KT) * NDIM * KTP + (long)c * NDIM * KTP;
  for (int n = threadIdx.x; n < NDIM; n += NTHREADS) {
    v8bf v0, v1, v2, v3;
#pragma unroll
    for (int k = 0; k < 8; ++k)  v0[k] = (bf16_t)Wa[(long)(k)      * NDIM + n];
#pragma unroll
    for (int k = 0; k < 8; ++k)  v1[k] = (bf16_t)Wa[(long)(k + 8)  * NDIM + n];
#pragma unroll
    for (int k = 0; k < 8; ++k)  v2[k] = (bf16_t)Wa[(long)(k + 16) * NDIM + n];
#pragma unroll
    for (int k = 0; k < 8; ++k)  v3[k] = (bf16_t)Wa[(long)(k + 24) * NDIM + n];
    bf16_t* d = da + (long)n * KTP;
    *(v8bf*)(d + 0)  = v0;
    *(v8bf*)(d + 8)  = v1;
    *(v8bf*)(d + 16) = v2;
    *(v8bf*)(d + 24) = v3;
  }
}

// --------------------------- fast hidden layer -----------------------------
// sAct[BM][HID] (bf16) x Wp (pre-converted chunk-major) -> relu(.+bias) -> sAct.
// Double-buffered async weight staging overlapped with WMMA.
template<int KD, bool RELU>
static __device__ __forceinline__ void layer_fast(const bf16_t* __restrict__ Wp,
                                                  const float* __restrict__ bg,
                                                  bf16_t* sAct, bf16_t* sW0, bf16_t* sW1,
                                                  int tid) {
  constexpr int NC = KD / KT;
  const int wave  = tid >> 5;
  const int lane  = tid & 31;
  const int l15   = lane & 15;
  const int khalf = lane >> 4;
  const int mB    = wave & 3;          // 4 M-blocks of 16 rows
  const int nb0   = (wave >> 2) * 16;  // 2 stripes x 16 n-blocks (256 cols)

  v8f acc[16];
#pragma unroll
  for (int i = 0; i < 16; ++i)
#pragma unroll
    for (int r = 0; r < 8; ++r) acc[i][r] = 0.0f;

  const bf16_t* aBase = sAct + (mB * 16 + l15) * HID + khalf * 8;

  stage_chunk(Wp, sW0, tid);                       // prefetch chunk 0
  for (int c = 0; c < NC; ++c) {
    bf16_t* cur = (c & 1) ? sW1 : sW0;
    bf16_t* nxt = (c & 1) ? sW0 : sW1;
    __syncthreads();                               // everyone done with chunk c-1 buffer
    if (c + 1 < NC) {
      stage_chunk(Wp + (long)(c + 1) * HID * KTP, nxt, tid);
#if USE_ASYNC_LDS
      __builtin_amdgcn_s_wait_asynccnt(CHUNK_OPS); // oldest 10 (= chunk c) retired
#endif
    } else {
#if USE_ASYNC_LDS
      __builtin_amdgcn_s_wait_asynccnt(0);
#endif
    }
    __syncthreads();                               // chunk c visible to all waves
    v16bf afrag = load_frag_a(aBase + c * KT);
#pragma unroll
    for (int i = 0; i < 16; ++i) {
      v16bf bfrag = load_frag_b(cur + ((nb0 + i) * 16 + l15) * KTP + khalf * 8);
      acc[i] = __builtin_amdgcn_wmma_f32_16x16x32_bf16(
          false, afrag, false, bfrag, (short)0, acc[i], false, false);
    }
  }
  __syncthreads();   // all A-reads of sAct done before overwrite

#pragma unroll
  for (int i = 0; i < 16; ++i) {
    const int col  = (nb0 + i) * 16 + l15;
    const float bias = bg[col];
#pragma unroll
    for (int r = 0; r < 8; ++r) {
      float v = acc[i][r] + bias;
      if (RELU) v = fmaxf(v, 0.0f);
      sAct[(mB * 16 + khalf * 8 + r) * HID + col] = (bf16_t)v;
    }
  }
  __syncthreads();   // outputs visible before next layer reads them
}

// ------------------------------ fast kernel --------------------------------
__global__ __launch_bounds__(NTHREADS)
void mlp_agents_fast(const float* __restrict__ state_seq,
                     const bf16_t* __restrict__ wsW1, const bf16_t* __restrict__ wsW2,
                     const bf16_t* __restrict__ wsW3, const bf16_t* __restrict__ wsW4,
                     const float* __restrict__ b1, const float* __restrict__ b2,
                     const float* __restrict__ b3, const float* __restrict__ b4,
                     float* __restrict__ out) {
  extern __shared__ bf16_t smem[];
  bf16_t* sAct = smem;                           // [BM][HID]      64 KB
  bf16_t* sW0  = smem + BM * HID;                // [HID][KTP]     40 KB
  bf16_t* sW1  = sW0 + HID * KTP;                // [HID][KTP]     40 KB

  const int  tid   = threadIdx.x;
  const int  agent = blockIdx.y;
  const long mBase = (long)blockIdx.x * BM;

  // Stage x = state_seq[:, SEQ-1, :] into sAct (bf16), coalesced float4 reads.
  {
    constexpr int GROUPS = DIN / 4;
    const int r0 = tid / GROUPS;
    const int c  = (tid % GROUPS) * 4;
    for (int r = r0; r < BM; r += NTHREADS / GROUPS) {
      const long m = mBase + r;
      v4f x = *(const v4f*)(state_seq + (m * SEQ + (SEQ - 1)) * DIN + c);
      bf16_t* dst = sAct + r * HID + c;
      dst[0] = (bf16_t)x[0];
      dst[1] = (bf16_t)x[1];
      dst[2] = (bf16_t)x[2];
      dst[3] = (bf16_t)x[3];
    }
  }
  // (barriers inside layer_fast cover this staging)

  const bf16_t* W1a = wsW1 + (long)agent * (DIN / KT) * HID * KTP;
  const bf16_t* W2a = wsW2 + (long)agent * (HID / KT) * HID * KTP;
  const bf16_t* W3a = wsW3 + (long)agent * (HID / KT) * HID * KTP;
  const bf16_t* W4a = wsW4 + (long)agent * (HID / KT) * KOUT * KTP;

  layer_fast<DIN, true>(W1a, b1 + (long)agent * HID, sAct, sW0, sW1, tid);
  layer_fast<HID, true>(W2a, b2 + (long)agent * HID, sAct, sW0, sW1, tid);
  layer_fast<HID, true>(W3a, b3 + (long)agent * HID, sAct, sW0, sW1, tid);

  // ---- Layer 4: [BM x 512] @ [512 x 16] + b4 -> global (f32) ----
  // B-fragments straight from L2 (tiny; reused by all 16 M-tiles per agent).
  const int wave  = tid >> 5;
  const int lane  = tid & 31;
  const int l15   = lane & 15;
  const int khalf = lane >> 4;

  if (wave < 4) {  // wave-uniform: EXEC all-ones inside (WMMA requirement)
    v8f acc;
#pragma unroll
    for (int r = 0; r < 8; ++r) acc[r] = 0.0f;
#pragma unroll
    for (int c = 0; c < HID / KT; ++c) {
      v16bf afrag = load_frag_a(sAct + (wave * 16 + l15) * HID + c * KT + khalf * 8);
      v16bf bfrag = load_frag_b(W4a + ((long)c * KOUT + l15) * KTP + khalf * 8);
      acc = __builtin_amdgcn_wmma_f32_16x16x32_bf16(
          false, afrag, false, bfrag, (short)0, acc, false, false);
    }
    const float bias = b4[(long)agent * KOUT + l15];
#pragma unroll
    for (int r = 0; r < 8; ++r) {
      const long m = mBase + wave * 16 + khalf * 8 + r;
      out[(m * NAGENT + agent) * KOUT + l15] = acc[r] + bias;
    }
  }
}

// ======================= fallback (no-workspace) path ======================
template<int NDIM>
static __device__ __forceinline__ void stage_weights_f32(const float* __restrict__ Wg,
                                                         int k0,
                                                         bf16_t* __restrict__ sWt,
                                                         int tid) {
  constexpr int GROUPS = NDIM / 4;
  constexpr int RSTEP  = NTHREADS / GROUPS;
  const int r0 = tid / GROUPS;
  const int c  = (tid % GROUPS) * 4;
  for (int r = r0; r < KT; r += RSTEP) {
    v4f w = *(const v4f*)(Wg + (long)(k0 + r) * NDIM + c);
    sWt[(c + 0) * KTP + r] = (bf16_t)w[0];
    sWt[(c + 1) * KTP + r] = (bf16_t)w[1];
    sWt[(c + 2) * KTP + r] = (bf16_t)w[2];
    sWt[(c + 3) * KTP + r] = (bf16_t)w[3];
  }
}

template<int KD, bool RELU>
static __device__ __forceinline__ void layer_conv(const float* __restrict__ Wg,
                                                  const float* __restrict__ bg,
                                                  bf16_t* sAct, bf16_t* sWt, int tid) {
  const int wave = tid >> 5, lane = tid & 31, l15 = lane & 15, khalf = lane >> 4;
  const int mB = wave & 3, nb0 = (wave >> 2) * 16;
  v8f acc[16];
#pragma unroll
  for (int i = 0; i < 16; ++i)
#pragma unroll
    for (int r = 0; r < 8; ++r) acc[i][r] = 0.0f;
  const bf16_t* aBase = sAct + (mB * 16 + l15) * HID + khalf * 8;
  for (int kc = 0; kc < KD; kc += KT) {
    __syncthreads();
    stage_weights_f32<HID>(Wg, kc, sWt, tid);
    __syncthreads();
    v16bf afrag = load_frag_a(aBase + kc);
#pragma unroll
    for (int i = 0; i < 16; ++i) {
      v16bf bfrag = load_frag_b(sWt + ((nb0 + i) * 16 + l15) * KTP + khalf * 8);
      acc[i] = __builtin_amdgcn_wmma_f32_16x16x32_bf16(
          false, afrag, false, bfrag, (short)0, acc[i], false, false);
    }
  }
  __syncthreads();
#pragma unroll
  for (int i = 0; i < 16; ++i) {
    const int col = (nb0 + i) * 16 + l15;
    const float bias = bg[col];
#pragma unroll
    for (int r = 0; r < 8; ++r) {
      float v = acc[i][r] + bias;
      if (RELU) v = fmaxf(v, 0.0f);
      sAct[(mB * 16 + khalf * 8 + r) * HID + col] = (bf16_t)v;
    }
  }
  __syncthreads();
}

__global__ __launch_bounds__(NTHREADS)
void mlp_agents_conv(const float* __restrict__ state_seq,
                     const float* __restrict__ W1, const float* __restrict__ b1,
                     const float* __restrict__ W2, const float* __restrict__ b2,
                     const float* __restrict__ W3, const float* __restrict__ b3,
                     const float* __restrict__ W4, const float* __restrict__ b4,
                     float* __restrict__ out) {
  extern __shared__ bf16_t smem[];
  bf16_t* sAct = smem;
  bf16_t* sWt  = smem + BM * HID;
  const int  tid   = threadIdx.x;
  const int  agent = blockIdx.y;
  const long mBase = (long)blockIdx.x * BM;
  {
    constexpr int GROUPS = DIN / 4;
    const int r0 = tid / GROUPS;
    const int c  = (tid % GROUPS) * 4;
    for (int r = r0; r < BM; r += NTHREADS / GROUPS) {
      const long m = mBase + r;
      v4f x = *(const v4f*)(state_seq + (m * SEQ + (SEQ - 1)) * DIN + c);
      bf16_t* dst = sAct + r * HID + c;
      dst[0] = (bf16_t)x[0]; dst[1] = (bf16_t)x[1];
      dst[2] = (bf16_t)x[2]; dst[3] = (bf16_t)x[3];
    }
  }
  layer_conv<DIN, true>(W1 + (long)agent * DIN * HID, b1 + (long)agent * HID, sAct, sWt, tid);
  layer_conv<HID, true>(W2 + (long)agent * HID * HID, b2 + (long)agent * HID, sAct, sWt, tid);
  layer_conv<HID, true>(W3 + (long)agent * HID * HID, b3 + (long)agent * HID, sAct, sWt, tid);

  const int wave = tid >> 5, lane = tid & 31, l15 = lane & 15, khalf = lane >> 4;
  v8f acc;
#pragma unroll
  for (int r = 0; r < 8; ++r) acc[r] = 0.0f;
  const float* W4a = W4 + (long)agent * HID * KOUT;
  for (int kc = 0; kc < HID; kc += KT) {
    __syncthreads();
    {
      constexpr int GROUPS = KOUT / 4;
      const int r0 = tid / GROUPS;
      const int c  = (tid % GROUPS) * 4;
      for (int r = r0; r < KT; r += NTHREADS / GROUPS) {
        v4f w = *(const v4f*)(W4a + (long)(kc + r) * KOUT + c);
        sWt[(c + 0) * KTP + r] = (bf16_t)w[0];
        sWt[(c + 1) * KTP + r] = (bf16_t)w[1];
        sWt[(c + 2) * KTP + r] = (bf16_t)w[2];
        sWt[(c + 3) * KTP + r] = (bf16_t)w[3];
      }
    }
    __syncthreads();
    if (wave < 4) {
      v16bf afrag = load_frag_a(sAct + (wave * 16 + l15) * HID + kc + khalf * 8);
      v16bf bfrag = load_frag_b(sWt + l15 * KTP + khalf * 8);
      acc = __builtin_amdgcn_wmma_f32_16x16x32_bf16(
          false, afrag, false, bfrag, (short)0, acc, false, false);
    }
  }
  if (wave < 4) {
    const float bias = b4[(long)agent * KOUT + l15];
#pragma unroll
    for (int r = 0; r < 8; ++r) {
      const long m = mBase + wave * 16 + khalf * 8 + r;
      out[(m * NAGENT + agent) * KOUT + l15] = acc[r] + bias;
    }
  }
}

// -------------------------------- host side --------------------------------
extern "C" void kernel_launch(void* const* d_in, const int* in_sizes, int n_in,
                              void* d_out, int out_size, void* d_ws, size_t ws_size,
                              hipStream_t stream) {
  const float* state = (const float*)d_in[0];
  // d_in[1] = actions_seq, unused by the reference computation
  const float* W1 = (const float*)d_in[2];
  const float* b1 = (const float*)d_in[3];
  const float* W2 = (const float*)d_in[4];
  const float* b2 = (const float*)d_in[5];
  const float* W3 = (const float*)d_in[6];
  const float* b3 = (const float*)d_in[7];
  const float* W4 = (const float*)d_in[8];
  const float* b4 = (const float*)d_in[9];
  float* out = (float*)d_out;

  const size_t elW1 = (size_t)NAGENT * (DIN / KT) * HID * KTP;   // 10.5M elems
  const size_t elW2 = (size_t)NAGENT * (HID / KT) * HID * KTP;   // 21.0M elems
  const size_t elW4 = (size_t)NAGENT * (HID / KT) * KOUT * KTP;  // 0.66M elems
  const size_t neededBytes = (elW1 + 2 * elW2 + elW4) * sizeof(bf16_t);  // ~101 MB

  const dim3 grid(NB / BM, NAGENT);  // 16 x 64 = 1024 blocks

  if (d_ws != nullptr && ws_size >= neededBytes) {
    bf16_t* wsW1 = (bf16_t*)d_ws;
    bf16_t* wsW2 = wsW1 + elW1;
    bf16_t* wsW3 = wsW2 + elW2;
    bf16_t* wsW4 = wsW3 + elW2;

    convert_weights_kernel<DIN, HID><<<dim3(DIN / KT, NAGENT), NTHREADS, 0, stream>>>(W1, wsW1);
    convert_weights_kernel<HID, HID><<<dim3(HID / KT, NAGENT), NTHREADS, 0, stream>>>(W2, wsW2);
    convert_weights_kernel<HID, HID><<<dim3(HID / KT, NAGENT), NTHREADS, 0, stream>>>(W3, wsW3);
    convert_weights_kernel<HID, KOUT><<<dim3(HID / KT, NAGENT), NTHREADS, 0, stream>>>(W4, wsW4);

    const size_t smem = (size_t)(BM * HID + 2 * HID * KTP) * sizeof(bf16_t);  // 144 KB
    mlp_agents_fast<<<grid, NTHREADS, smem, stream>>>(
        state, wsW1, wsW2, wsW3, wsW4, b1, b2, b3, b4, out);
  } else {
    const size_t smem = (size_t)(BM * HID + HID * KTP) * sizeof(bf16_t);      // 104 KB
    mlp_agents_conv<<<grid, NTHREADS, smem, stream>>>(
        state, W1, b1, W2, b2, W3, b3, W4, b4, out);
  }

  (void)in_sizes; (void)n_in; (void)out_size;
}
